// ChebEncoder_4853313045127
// MI455X (gfx1250) — matrix operand order repaired and checked by reading, not compiled, not run on hardware
//
#include <hip/hip_runtime.h>
#include <hip/hip_bf16.h>
#include <stdint.h>

#define N_NODES 50000
#define N_EDGES 160000
#define DIM     512
#define M_TILES (N_NODES / 16)   // 3125, exact
#define K_TILES (DIM / 32)       // 16
#define N_TILES (DIM / 16)       // 32

typedef __attribute__((ext_vector_type(16))) __bf16 bf16x16;
typedef __attribute__((ext_vector_type(8)))  float  floatx8;
typedef __attribute__((ext_vector_type(4)))  unsigned int uint32x4;
typedef __attribute__((ext_vector_type(8)))  int int32x8;
typedef __attribute__((ext_vector_type(4)))  int int32x4;

union Frag32B { uint4 u[2]; bf16x16 v; };

#if defined(__has_builtin)
#if __has_builtin(__builtin_amdgcn_tensor_load_to_lds)
#define HAVE_TDM 1
#endif
#endif

static __device__ __forceinline__ unsigned short f2bf(float f) {
  union { float f; unsigned int u; } cv; cv.f = f;
  unsigned int u = cv.u;
  unsigned int r = 0x7FFFu + ((u >> 16) & 1u);   // round-to-nearest-even
  return (unsigned short)((u + r) >> 16);
}

// ---------------------------------------------------------------- utilities
__global__ void zero_f32(float* __restrict__ p, size_t n) {
  size_t i = (size_t)blockIdx.x * blockDim.x + threadIdx.x;
  size_t stride = (size_t)gridDim.x * blockDim.x;
  for (; i < n; i += stride) p[i] = 0.0f;
}

__global__ void cvt_f32_bf16(const float* __restrict__ in,
                             unsigned short* __restrict__ out, size_t n) {
  size_t i = (size_t)blockIdx.x * blockDim.x + threadIdx.x;
  size_t stride = (size_t)gridDim.x * blockDim.x;
  for (; i < n; i += stride) out[i] = f2bf(in[i]);
}

// ---------------------------------------------------------------- graph norm
__global__ void deg_kernel(const long long* __restrict__ src,
                           float* __restrict__ deg, int E) {
  int e = blockIdx.x * blockDim.x + threadIdx.x;
  if (e < E) atomicAdd(&deg[src[e]], 1.0f);
}

__global__ void dinv_kernel(const float* __restrict__ deg,
                            float* __restrict__ dinv, int n) {
  int i = blockIdx.x * blockDim.x + threadIdx.x;
  if (i < n) {
    float d = deg[i];
    dinv[i] = (d > 0.0f) ? rsqrtf(fmaxf(d, 1.0f)) : 0.0f;
  }
}

__global__ void norm_kernel(const long long* __restrict__ src,
                            const long long* __restrict__ dst,
                            const float* __restrict__ dinv,
                            float* __restrict__ nrm, int E) {
  int e = blockIdx.x * blockDim.x + threadIdx.x;
  if (e < E) nrm[e] = -dinv[src[e]] * dinv[dst[e]];
}

// scatter: tx[dst] += h[src] * norm  (one block per edge, 512 floats)
__global__ __launch_bounds__(256)
void scatter_prop(const float* __restrict__ feat,
                  const long long* __restrict__ src,
                  const long long* __restrict__ dst,
                  const float* __restrict__ nrm,
                  float* __restrict__ acc) {
  int e = blockIdx.x;
  long long s = src[e], d = dst[e];
  float w = nrm[e];
  const float* fs = feat + (size_t)s * DIM;
  float* fd = acc + (size_t)d * DIM;
  int c = threadIdx.x;
  atomicAdd(&fd[c],       fs[c]       * w);
  atomicAdd(&fd[c + 256], fs[c + 256] * w);
}

// -------------------------------------------------- weight packing (B layout)
// Packed layout [K_TILES][N_TILES][32 lanes][16 elems] bf16:
//   lane half = lane>>4, n = ntile*16 + (lane&15), k = ktile*32 + half*16 + j
__global__ __launch_bounds__(256)
void pack_weight(const float* __restrict__ W, unsigned short* __restrict__ P) {
  int idx = blockIdx.x * blockDim.x + threadIdx.x;      // 0 .. 262143
  int j    = idx & 15;
  int lane = (idx >> 4) & 31;
  int nt   = (idx >> 9) & 31;
  int kt   = idx >> 14;
  int half = lane >> 4;
  int k = kt * 32 + half * 16 + j;
  int n = nt * 16 + (lane & 15);
  P[idx] = f2bf(W[(size_t)k * DIM + n]);
}

// --------------------------------------------------------- TDM slab load
// D# per CDNA5 ISA ch.10/§8: group0 = count|lds_addr|global_addr|type=2,
// group1 = data_size=2B, tensor 512x16 (== tile, so never OOB), stride 512.
#ifdef HAVE_TDM
static __device__ __forceinline__ void tdm_load_tile(const unsigned short* gsrc,
                                                     unsigned int ldsOff) {
  unsigned long long ga = (unsigned long long)(uintptr_t)gsrc;
  uint32x4 g0;
  g0.x = 1u;                                   // count=1 (valid), user mode
  g0.y = ldsOff;                               // lds_addr (bytes)
  g0.z = (unsigned int)(ga & 0xFFFFFFFFull);   // global_addr[31:0]
  g0.w = (unsigned int)((ga >> 32) & 0x01FFFFFFull) | 0x80000000u; // [56:32]|type=2
  int32x8 g1;
  g1[0] = 0x00010000;    // workgroup_mask=0, data_size=1 (2 bytes)
  g1[1] = 512 << 16;     // tensor_dim0 = 512   (bits 63:48)
  g1[2] = 16  << 16;     // tensor_dim1 = 16    (bits 111:96... lo16)
  g1[3] = 512 << 16;     // tile_dim0   = 512   (bits 127:112)
  g1[4] = 16;            // tile_dim1   = 16, tile_dim2 = 0
  g1[5] = 512;           // tensor_dim0_stride = 512 (lo32)
  g1[6] = 0;
  g1[7] = 0;
  int32x4 z4 = {0, 0, 0, 0};
#if __clang_major__ >= 23
  int32x8 z8 = {0, 0, 0, 0, 0, 0, 0, 0};
  __builtin_amdgcn_tensor_load_to_lds(g0, g1, z4, z4, z8, 0);
#else
  __builtin_amdgcn_tensor_load_to_lds(g0, g1, z4, z4, 0);
#endif
}
#endif

// ------------------------------------------------------------------- GEMM
// out[M,512] = A0 @ W0 + A1 @ W1 + bias   (A bf16 row-major staged in LDS by
// TDM, W packed bf16 in L2, f32 accumulate).  Block = 8 waves; wave w owns
// ntiles [4w,4w+4); block owns 16 rows (one mtile) x 512 cols.
__global__ __launch_bounds__(256)
void gemm_cheb(const unsigned short* __restrict__ A0,
               const unsigned short* __restrict__ A1,
               const unsigned short* __restrict__ B0p,
               const unsigned short* __restrict__ B1p,
               const float* __restrict__ bias,
               float* __restrict__ out) {
  __shared__ __align__(16) unsigned short Atile[2][16 * DIM];  // 2 x 16 KB

  const int mtile = blockIdx.x;
  const int wave  = threadIdx.x >> 5;
  const int lane  = threadIdx.x & 31;
  const int r     = lane & 15;
  const int half  = lane >> 4;
  const size_t slab = (size_t)mtile * 16 * DIM;

#ifdef HAVE_TDM
  if (threadIdx.x < 32) {   // one wave issues the DMAs; TDM ignores EXEC
    unsigned int lds0 = (unsigned int)(uintptr_t)&Atile[0][0];
    unsigned int lds1 = (unsigned int)(uintptr_t)&Atile[1][0];
    tdm_load_tile(A0 + slab, lds0);
    tdm_load_tile(A1 + slab, lds1);
    __builtin_amdgcn_s_wait_tensorcnt(0);
  }
  __syncthreads();
#else
  {
    // cooperative fallback: 256 threads copy 2 x 16 KB as b128
    const uint4* g0 = (const uint4*)(A0 + slab);
    const uint4* g1 = (const uint4*)(A1 + slab);
    uint4* l0 = (uint4*)&Atile[0][0];
    uint4* l1 = (uint4*)&Atile[1][0];
    for (int i = threadIdx.x; i < (16 * DIM) / 8; i += 256) {
      l0[i] = g0[i];
      l1[i] = g1[i];
    }
    __syncthreads();
  }
#endif

  floatx8 acc0 = {0,0,0,0,0,0,0,0};
  floatx8 acc1 = {0,0,0,0,0,0,0,0};
  floatx8 acc2 = {0,0,0,0,0,0,0,0};
  floatx8 acc3 = {0,0,0,0,0,0,0,0};

  for (int s = 0; s < 2; ++s) {
    const unsigned short* Ar = &Atile[s][(unsigned)r * DIM];
    const unsigned short* Bw = s ? B1p : B0p;
    for (int kt = 0; kt < K_TILES; ++kt) {
      const int kb = kt * 32;
      // A fragment: 16x32 bf16; lane (r,half) holds K = {half*16..+15}
      Frag32B af;
      af.u[0] = *(const uint4*)(Ar + kb + half * 16);
      af.u[1] = *(const uint4*)(Ar + kb + half * 16 + 8);
      const unsigned short* bbase =
          Bw + (((size_t)kt * N_TILES + wave * 4) * 32 + lane) * 16;
#pragma unroll
      for (int a = 0; a < 4; ++a) {
        Frag32B bf;
        const unsigned short* bp = bbase + (size_t)a * 32 * 16;
        bf.u[0] = *(const uint4*)(bp);
        bf.u[1] = *(const uint4*)(bp + 8);
        floatx8 c = (a == 0) ? acc0 : (a == 1) ? acc1 : (a == 2) ? acc2 : acc3;
        c = __builtin_amdgcn_wmma_f32_16x16x32_bf16(
                false, af.v, false, bf.v, (short)0, c, false, false);
        if      (a == 0) acc0 = c;
        else if (a == 1) acc1 = c;
        else if (a == 2) acc2 = c;
        else             acc3 = c;
      }
    }
  }

  // epilogue: C/D layout -> row = i + 8*half, col = lane&15 within each ntile
#pragma unroll
  for (int a = 0; a < 4; ++a) {
    const floatx8 c = (a == 0) ? acc0 : (a == 1) ? acc1 : (a == 2) ? acc2 : acc3;
    const int col = (wave * 4 + a) * 16 + r;
    const float bc = bias[col];
#pragma unroll
    for (int i = 0; i < 8; ++i) {
      const int row = mtile * 16 + i + half * 8;
      out[(size_t)row * DIM + col] = c[i] + bc;
    }
  }
}

// ------------------------------------------------- LayerNorm + LeakyReLU
__global__ __launch_bounds__(256)
void ln_lrelu(float* __restrict__ h, unsigned short* __restrict__ hb,
              const float* __restrict__ gamma, const float* __restrict__ beta) {
  __shared__ float ssum[256];
  __shared__ float ssq[256];
  const int row = blockIdx.x;
  const int t = threadIdx.x;
  float* hr = h + (size_t)row * DIM;
  float v0 = hr[t];
  float v1 = hr[t + 256];
  ssum[t] = v0 + v1;
  ssq[t]  = v0 * v0 + v1 * v1;
  __syncthreads();
  for (int s = 128; s > 0; s >>= 1) {
    if (t < s) { ssum[t] += ssum[t + s]; ssq[t] += ssq[t + s]; }
    __syncthreads();
  }
  const float mu  = ssum[0] * (1.0f / DIM);
  const float var = ssq[0] * (1.0f / DIM) - mu * mu;
  const float rs  = rsqrtf(var + 1e-5f);

  float y0 = (v0 - mu) * rs * gamma[t] + beta[t];
  float y1 = (v1 - mu) * rs * gamma[t + 256] + beta[t + 256];
  y0 = (y0 >= 0.0f) ? y0 : 0.01f * y0;
  y1 = (y1 >= 0.0f) ? y1 : 0.01f * y1;
  hr[t]       = y0;
  hr[t + 256] = y1;
  unsigned short* hbr = hb + (size_t)row * DIM;
  hbr[t]       = f2bf(y0);
  hbr[t + 256] = f2bf(y1);
}

// -------------------------------------------------------------------- launch
extern "C" void kernel_launch(void* const* d_in, const int* in_sizes, int n_in,
                              void* d_out, int out_size, void* d_ws, size_t ws_size,
                              hipStream_t stream) {
  const float* x     = (const float*)d_in[0];
  const float* W1_0  = (const float*)d_in[1];
  const float* W1_1  = (const float*)d_in[2];
  const float* b1    = (const float*)d_in[3];
  const float* gamma = (const float*)d_in[4];
  const float* beta  = (const float*)d_in[5];
  const float* W2_0  = (const float*)d_in[6];
  const float* W2_1  = (const float*)d_in[7];
  const float* b2    = (const float*)d_in[8];
  const long long* ei  = (const long long*)d_in[9];
  const long long* src = ei;
  const long long* dst = ei + N_EDGES;
  float* out = (float*)d_out;

  const size_t ND = (size_t)N_NODES * DIM;
  char* ws = (char*)d_ws;
  size_t off = 0;
  auto carve = [&](size_t bytes) -> char* {
    char* p = ws + off;
    off += (bytes + 255) & ~(size_t)255;
    return p;
  };
  float* deg  = (float*)carve((size_t)N_NODES * 4);
  float* dinv = (float*)carve((size_t)N_NODES * 4);
  float* nrm  = (float*)carve((size_t)N_EDGES * 4);
  unsigned short* Wp10 = (unsigned short*)carve((size_t)DIM * DIM * 2);
  unsigned short* Wp11 = (unsigned short*)carve((size_t)DIM * DIM * 2);
  unsigned short* Wp20 = (unsigned short*)carve((size_t)DIM * DIM * 2);
  unsigned short* Wp21 = (unsigned short*)carve((size_t)DIM * DIM * 2);
  unsigned short* xb   = (unsigned short*)carve(ND * 2);   // layer-1 A; reused as hb
  unsigned short* txb  = (unsigned short*)carve(ND * 2);
  float* txf = (float*)carve(ND * 4);
  float* h   = (float*)carve(ND * 4);

  const int T = 256;
  const int gridE = (N_EDGES + T - 1) / T;
  const int gridN = (N_NODES + T - 1) / T;
  const int gridND = (int)((ND + T - 1) / T);
  const int gridPack = (DIM * DIM) / T;

  // ---- graph normalization ------------------------------------------------
  zero_f32<<<gridN, T, 0, stream>>>(deg, (size_t)N_NODES);
  zero_f32<<<gridND, T, 0, stream>>>(txf, ND);
  deg_kernel<<<gridE, T, 0, stream>>>(src, deg, N_EDGES);
  dinv_kernel<<<gridN, T, 0, stream>>>(deg, dinv, N_NODES);
  norm_kernel<<<gridE, T, 0, stream>>>(src, dst, dinv, nrm, N_EDGES);

  // ---- weight packing (bf16, WMMA-B layout) -------------------------------
  pack_weight<<<gridPack, T, 0, stream>>>(W1_0, Wp10);
  pack_weight<<<gridPack, T, 0, stream>>>(W1_1, Wp11);
  pack_weight<<<gridPack, T, 0, stream>>>(W2_0, Wp20);
  pack_weight<<<gridPack, T, 0, stream>>>(W2_1, Wp21);

  // ---- layer 1 ------------------------------------------------------------
  cvt_f32_bf16<<<gridND, T, 0, stream>>>(x, xb, ND);
  scatter_prop<<<N_EDGES, T, 0, stream>>>(x, src, dst, nrm, txf);
  cvt_f32_bf16<<<gridND, T, 0, stream>>>(txf, txb, ND);
  gemm_cheb<<<M_TILES, T, 0, stream>>>(xb, txb, Wp10, Wp11, b1, h);

  // ---- LayerNorm + LeakyReLU (in place; bf16 copy into recycled xb) -------
  ln_lrelu<<<N_NODES, T, 0, stream>>>(h, xb, gamma, beta);

  // ---- layer 2 ------------------------------------------------------------
  zero_f32<<<gridND, T, 0, stream>>>(txf, ND);
  scatter_prop<<<N_EDGES, T, 0, stream>>>(h, src, dst, nrm, txf);
  cvt_f32_bf16<<<gridND, T, 0, stream>>>(txf, txb, ND);
  gemm_cheb<<<M_TILES, T, 0, stream>>>(xb, txb, Wp20, Wp21, b2, out);
}
